// SelfAttention_48249662603402
// MI455X (gfx1250) — matrix operand order repaired and checked by reading, not compiled, not run on hardware
//
#include <hip/hip_runtime.h>
#include <hip/hip_bf16.h>

// ---------------- problem constants ----------------
constexpr int kB       = 4;
constexpr int kN       = 2048;
constexpr int kDim     = 768;
constexpr int kHeads   = 12;
constexpr int kHd      = 64;            // head dim
constexpr int kKvDim   = 2 * kDim;      // 1536
constexpr int kM       = kB * kN;       // 8192 rows
constexpr float kScale = 0.125f;        // 64^-0.5

// ---------------- types ----------------
typedef __bf16 bf16;
typedef __attribute__((ext_vector_type(16))) __bf16 v16bf;
typedef __attribute__((ext_vector_type(8)))  __bf16 v8bf;
typedef __attribute__((ext_vector_type(4)))  __bf16 v4bf;
typedef __attribute__((ext_vector_type(8)))  float  v8f;
typedef __attribute__((ext_vector_type(4)))  float  v4f;
typedef __attribute__((ext_vector_type(4)))  unsigned int u32x4;
typedef __attribute__((ext_vector_type(8)))  int          i32x8;
typedef __attribute__((ext_vector_type(4)))  int          i32x4;

#define WMMA_BF16(a, b, c) \
  __builtin_amdgcn_wmma_f32_16x16x32_bf16(false, (a), false, (b), (short)0, (c), false, false)

#if __has_builtin(__builtin_amdgcn_tensor_load_to_lds) && \
    __has_builtin(__builtin_amdgcn_s_wait_tensorcnt)
#define HAVE_TDM 1
#else
#define HAVE_TDM 0
#endif

// ---------------- fragment helpers ----------------
__device__ __forceinline__ v16bf cat16(v8bf lo, v8bf hi) {
  v16bf r;
#pragma unroll
  for (int i = 0; i < 8; ++i) { r[i] = lo[i]; r[i + 8] = hi[i]; }
  return r;
}
// B-operand fragment: 16 contiguous bf16 (lane = N column, elems = K)
__device__ __forceinline__ v16bf ld_frag_b(const bf16* p) {
  return cat16(*(const v8bf*)p, *(const v8bf*)(p + 8));
}
// A-operand fragment: per-lane K elems are {k..k+15} (contiguous)
__device__ __forceinline__ v16bf ld_frag_a(const bf16* p) {
  return cat16(*(const v8bf*)p, *(const v8bf*)(p + 8));
}

// ---------------- VALU 16-lane butterfly reductions ----------------
__device__ __forceinline__ float permx16(float v, unsigned lo, unsigned hi) {
#if __has_builtin(__builtin_amdgcn_permlane16)
  unsigned u = __builtin_bit_cast(unsigned, v);
  u = __builtin_amdgcn_permlane16(u, u, lo, hi, false, false);
  return __builtin_bit_cast(float, u);
#else
  (void)hi;
  int mask = (lo == 0x67452301u) ? 1 : (lo == 0x45670123u) ? 2
           : (lo == 0x01234567u) ? 4 : 8;
  return __shfl_xor(v, mask);
#endif
}
__device__ __forceinline__ float rowmax16(float v) {
  v = fmaxf(v, permx16(v, 0x67452301u, 0xEFCDAB89u));  // xor 1
  v = fmaxf(v, permx16(v, 0x45670123u, 0xCDEF89ABu));  // xor 2
  v = fmaxf(v, permx16(v, 0x01234567u, 0x89ABCDEFu));  // xor 4
  v = fmaxf(v, permx16(v, 0xFEDCBA98u, 0x76543210u));  // xor 8
  return v;
}
__device__ __forceinline__ float rowsum16(float v) {
  v += permx16(v, 0x67452301u, 0xEFCDAB89u);
  v += permx16(v, 0x45670123u, 0xCDEF89ABu);
  v += permx16(v, 0x01234567u, 0x89ABCDEFu);
  v += permx16(v, 0xFEDCBA98u, 0x76543210u);
  return v;
}

// ---------------- Tensor Data Mover: 2D tile -> LDS ----------------
// Loads tile_d1 rows x tile_d0 bf16 elements, row pitch stride_elems, into LDS
// at lds_off with pad_amount DWORDs appended every pad_interval DWORDs
// (codes per D# group1: interval 0=2,1=4,2=8,3=16,4=32; amount 0=1,...,3=4).
#if HAVE_TDM
__device__ __forceinline__ void tdm_load_2d(unsigned lds_off, const void* gsrc,
                                            unsigned tile_d0, unsigned tile_d1,
                                            unsigned stride_elems,
                                            unsigned pad_int_code,
                                            unsigned pad_amt_code) {
  unsigned long long ga = (unsigned long long)(uintptr_t)gsrc;
  u32x4 g0;
  g0[0] = 1u;                                             // count=1, user D#
  g0[1] = lds_off;                                        // lds_addr (bytes)
  g0[2] = (unsigned)ga;                                   // global_addr[31:0]
  g0[3] = (unsigned)((ga >> 32) & 0x01ffffffu) | 0x80000000u; // [56:32] | type=2
  i32x8 g1;
  g1[0] = (int)((1u << 16) | (1u << 20) |                 // data_size=2B, pad_en
                (pad_int_code << 22) | (pad_amt_code << 25));
  g1[1] = (int)((tile_d0 & 0xffffu) << 16);               // tensor_dim0 lo16
  g1[2] = (int)((tile_d1 & 0xffffu) << 16);               // dim0 hi=0 | tensor_dim1 lo16
  g1[3] = (int)((tile_d0 & 0xffffu) << 16);               // dim1 hi=0 | tile_dim0
  g1[4] = (int)(tile_d1 & 0xffffu);                       // tile_dim1 | tile_dim2=0
  g1[5] = (int)stride_elems;                              // tensor_dim0_stride lo32
  g1[6] = 0;                                              // stride hi | dim1_stride lo
  g1[7] = 0;
  i32x4 z4 = {0, 0, 0, 0};
  i32x8 z8 = {0, 0, 0, 0, 0, 0, 0, 0};
  // 6-arg toolchain form: (g0, g1, g2, g3, extra, cpol)
  __builtin_amdgcn_tensor_load_to_lds(g0, g1, z4, z4, z8, 0);
}
__device__ __forceinline__ unsigned lds_off_of(const void* p) {
  return (unsigned)(uintptr_t)p;
}
#endif

// ---------------- stage 0: f32 -> bf16 ----------------
__global__ void cast_f32_to_bf16(const float* __restrict__ in,
                                 bf16* __restrict__ out, int n4) {
  int i = blockIdx.x * blockDim.x + threadIdx.x;
  if (i >= n4) return;
  v4f v = *(const v4f*)(in + 4 * (size_t)i);
  v4bf o;
#pragma unroll
  for (int k = 0; k < 4; ++k) o[k] = (bf16)v[k];
  *(v4bf*)(out + 4 * (size_t)i) = o;
}

// ---------------- stage 1/3: C = A(MxK) @ W(NoutxK)^T ----------------
// Block tile 128(M) x 64(N), BK = 32. 8 waves, each a 32x32 subtile as 2x2
// v_wmma_f32_16x16x32_bf16. Tiles double-buffered; staged by TDM when present.
template <bool OUT_F32>
__global__ __launch_bounds__(256)
void gemm_xwt(const bf16* __restrict__ A, const bf16* __restrict__ W,
              float* __restrict__ Cf, bf16* __restrict__ Cb,
              int M, int Nout, int K) {
  __shared__ bf16 As[2][128][40];   // 32-elem rows + 4-DWORD pad
  __shared__ bf16 Bs[2][64][40];

  const int tid  = threadIdx.x;
  const int lane = tid & 31;
  const int wave = tid >> 5;
  const int wm   = wave >> 1;
  const int wn   = wave & 1;
  const int mblk = blockIdx.y * 128;
  const int nblk = blockIdx.x * 64;
  const int mr   = lane & 15;
  const int kh   = lane >> 4;

  v8f c00 = {}, c01 = {}, c10 = {}, c11 = {};
  const int nk = K / 32;

#if HAVE_TDM
  if (wave == 0) {
    tdm_load_2d(lds_off_of(&As[0][0][0]), A + (size_t)mblk * K, 32, 128, K, 3, 3);
    tdm_load_2d(lds_off_of(&Bs[0][0][0]), W + (size_t)nblk * K, 32, 64, K, 3, 3);
  }
#else
  const bf16* gA = A + (size_t)(mblk + (tid >> 1)) * K + (tid & 1) * 16;
  const bf16* gW = W + (size_t)(nblk + (tid >> 2)) * K + (tid & 3) * 8;
  v8bf ar0 = *(const v8bf*)(gA);
  v8bf ar1 = *(const v8bf*)(gA + 8);
  v8bf br0 = *(const v8bf*)(gW);
#endif

  for (int j = 0; j < nk; ++j) {
    const int cur = j & 1;
#if HAVE_TDM
    if (wave == 0) __builtin_amdgcn_s_wait_tensorcnt(0);
#else
    *(v8bf*)&As[cur][tid >> 1][(tid & 1) * 16 + 0] = ar0;
    *(v8bf*)&As[cur][tid >> 1][(tid & 1) * 16 + 8] = ar1;
    *(v8bf*)&Bs[cur][tid >> 2][(tid & 3) * 8]      = br0;
#endif
    __syncthreads();

    if (j + 1 < nk) {
      const int kb = (j + 1) * 32;
#if HAVE_TDM
      if (wave == 0) {
        tdm_load_2d(lds_off_of(&As[cur ^ 1][0][0]),
                    A + (size_t)mblk * K + kb, 32, 128, K, 3, 3);
        tdm_load_2d(lds_off_of(&Bs[cur ^ 1][0][0]),
                    W + (size_t)nblk * K + kb, 32, 64, K, 3, 3);
      }
#else
      ar0 = *(const v8bf*)(gA + kb);
      ar1 = *(const v8bf*)(gA + kb + 8);
      br0 = *(const v8bf*)(gW + kb);
      __builtin_prefetch(gA + kb + 32, 0, 1);
      __builtin_prefetch(gW + kb + 32, 0, 1);
#endif
    }

    v16bf a0 = ld_frag_a(&As[cur][wm * 32 +      mr][16 * kh]);
    v16bf a1 = ld_frag_a(&As[cur][wm * 32 + 16 + mr][16 * kh]);
    v16bf b0 = ld_frag_b(&Bs[cur][wn * 32 +      mr][16 * kh]);
    v16bf b1 = ld_frag_b(&Bs[cur][wn * 32 + 16 + mr][16 * kh]);
    c00 = WMMA_BF16(a0, b0, c00);
    c01 = WMMA_BF16(a0, b1, c01);
    c10 = WMMA_BF16(a1, b0, c10);
    c11 = WMMA_BF16(a1, b1, c11);
    __syncthreads();
  }

  // C layout: lane 0-15 -> col=lane, rows r; lane 16-31 -> col=lane-16, rows 8+r
  const int gm0 = mblk + wm * 32 + 8 * kh;
  const int gn0 = nblk + wn * 32 + mr;
#pragma unroll
  for (int r = 0; r < 8; ++r) {
    const size_t m0 = (size_t)(gm0 + r)      * Nout;
    const size_t m1 = (size_t)(gm0 + 16 + r) * Nout;
    if (OUT_F32) {
      Cf[m0 + gn0]      = c00[r];
      Cf[m0 + gn0 + 16] = c01[r];
      Cf[m1 + gn0]      = c10[r];
      Cf[m1 + gn0 + 16] = c11[r];
    } else {
      Cb[m0 + gn0]      = (bf16)c00[r];
      Cb[m0 + gn0 + 16] = (bf16)c01[r];
      Cb[m1 + gn0]      = (bf16)c10[r];
      Cb[m1 + gn0 + 16] = (bf16)c11[r];
    }
  }
}

// ---------------- stage 2: flash attention ----------------
// grid (N/64, H, B), 128 threads = 4 waves; wave w owns 16 query rows.
// K tiles via TDM (double-buffered), V tiles register-prefetched and stored
// transposed, online softmax with VALU permlane16 reductions.
__global__ __launch_bounds__(128)
void attn_fwd(const bf16* __restrict__ Q, const bf16* __restrict__ KV,
              bf16* __restrict__ O) {
  __shared__ bf16 Ks[2][32][72];     // [key][d], 64-elem rows + 4-DWORD pad
  __shared__ bf16 Vst[2][64][40];    // [d][key] transposed V tile
  __shared__ bf16 Pt[4][16][40];     // per-wave P staging [qrow][key]

  const int tid  = threadIdx.x;
  const int lane = tid & 31;
  const int wave = tid >> 5;
  const int mr   = lane & 15;
  const int kh   = lane >> 4;

  const int b  = blockIdx.z;
  const int h  = blockIdx.y;
  const int q0 = blockIdx.x * 64 + wave * 16;

  const size_t qrow = ((size_t)(b * kN) + q0 + mr) * kDim + h * kHd;
  const v16bf qa0 = ld_frag_a(Q + qrow +      16 * kh);
  const v16bf qa1 = ld_frag_a(Q + qrow + 32 + 16 * kh);

  float rmax[8], rsum[8];
  v8f o0 = {}, o1 = {}, o2 = {}, o3 = {};
#pragma unroll
  for (int r = 0; r < 8; ++r) { rmax[r] = -__builtin_inff(); rsum[r] = 0.f; }

  const int    krow   = tid >> 2;         // 0..31 key within tile
  const int    kseg   = (tid & 3) * 16;   // d offset handled by this thread
  const size_t kvbase = (size_t)(b * kN) * kKvDim + h * 128;
  const int    nt     = kN / 32;

  // V tile register prefetch (transposed into LDS at top of each iteration)
  v8bf vv0, vv1;
  {
    const bf16* vp = KV + kvbase + (size_t)krow * kKvDim + 64 + kseg;
    vv0 = *(const v8bf*)(vp);
    vv1 = *(const v8bf*)(vp + 8);
  }
#if HAVE_TDM
  if (wave == 0)
    tdm_load_2d(lds_off_of(&Ks[0][0][0]), KV + kvbase, 64, 32, kKvDim, 4, 3);
#else
  v8bf kr0, kr1;
  {
    const bf16* kp = KV + kvbase + (size_t)krow * kKvDim + kseg;
    kr0 = *(const v8bf*)(kp);
    kr1 = *(const v8bf*)(kp + 8);
  }
#endif

  for (int j = 0; j < nt; ++j) {
    const int cur = j & 1;
    const int jb  = j * 32;

    // ---- publish this tile to LDS ----
#pragma unroll
    for (int e = 0; e < 8; ++e) {
      Vst[cur][kseg + e][krow]     = vv0[e];
      Vst[cur][kseg + 8 + e][krow] = vv1[e];
    }
#if HAVE_TDM
    if (wave == 0) __builtin_amdgcn_s_wait_tensorcnt(0);
#else
    *(v8bf*)&Ks[cur][krow][kseg]     = kr0;
    *(v8bf*)&Ks[cur][krow][kseg + 8] = kr1;
#endif
    __syncthreads();

    // ---- prefetch next tile while computing this one ----
    if (j + 1 < nt) {
      const size_t nb = kvbase + (size_t)(jb + 32 + krow) * kKvDim;
      const bf16* vp = KV + nb + 64 + kseg;
      vv0 = *(const v8bf*)(vp);
      vv1 = *(const v8bf*)(vp + 8);
#if HAVE_TDM
      if (wave == 0)
        tdm_load_2d(lds_off_of(&Ks[cur ^ 1][0][0]),
                    KV + kvbase + (size_t)(jb + 32) * kKvDim, 64, 32, kKvDim, 4, 3);
#else
      const bf16* kp = KV + nb + kseg;
      kr0 = *(const v8bf*)(kp);
      kr1 = *(const v8bf*)(kp + 8);
#endif
    }

    // ---- scores: two 16x16 tiles over this 32-key block ----
    // Load ALL four B-fragments first (independent values) so the scheduler
    // can batch the 8 ds_load_b128s and overlap LDS latency with the WMMAs.
    const v16bf kb00 = ld_frag_b(&Ks[cur][     mr][     16 * kh]);
    const v16bf kb01 = ld_frag_b(&Ks[cur][     mr][32 + 16 * kh]);
    const v16bf kb10 = ld_frag_b(&Ks[cur][16 + mr][     16 * kh]);
    const v16bf kb11 = ld_frag_b(&Ks[cur][16 + mr][32 + 16 * kh]);
    v8f s0 = {}, s1 = {};
    s0 = WMMA_BF16(qa0, kb00, s0);
    s1 = WMMA_BF16(qa0, kb10, s1);
    s0 = WMMA_BF16(qa1, kb01, s0);
    s1 = WMMA_BF16(qa1, kb11, s1);

    float sa[8], sb[8];
#pragma unroll
    for (int r = 0; r < 8; ++r) {
      float xa = s0[r] * kScale;
      float xb = s1[r] * kScale;
      // reference masks attn[..., 1:, 0]: key 0 for every query > 0
      if (jb == 0 && mr == 0 && (q0 + 8 * kh + r) > 0) xa = -3.0e38f;
      sa[r] = xa;
      sb[r] = xb;
    }

    // ---- online softmax update (VALU butterflies) ----
#pragma unroll
    for (int r = 0; r < 8; ++r) {
      const float t    = rowmax16(fmaxf(sa[r], sb[r]));
      const float nm   = fmaxf(rmax[r], t);
      const float corr = __expf(rmax[r] - nm);
      rmax[r] = nm;
      const float pa = __expf(sa[r] - nm);
      const float pb = __expf(sb[r] - nm);
      rsum[r] = rsum[r] * corr + rowsum16(pa + pb);
      o0[r] *= corr; o1[r] *= corr; o2[r] *= corr; o3[r] *= corr;
      Pt[wave][8 * kh + r][mr]      = (bf16)pa;
      Pt[wave][8 * kh + r][16 + mr] = (bf16)pb;
    }
    asm volatile("" ::: "memory");   // LDS ops are in-order per wave; block reordering

    // ---- O += P @ V (K-dim = 32 keys, N-dim = 64 d in four 16-tiles) ----
    const v16bf pfrag = ld_frag_a(&Pt[wave][mr][16 * kh]);
    const v16bf vb0 = ld_frag_b(&Vst[cur][ 0 + mr][16 * kh]);
    const v16bf vb1 = ld_frag_b(&Vst[cur][16 + mr][16 * kh]);
    const v16bf vb2 = ld_frag_b(&Vst[cur][32 + mr][16 * kh]);
    const v16bf vb3 = ld_frag_b(&Vst[cur][48 + mr][16 * kh]);
    o0 = WMMA_BF16(pfrag, vb0, o0);
    o1 = WMMA_BF16(pfrag, vb1, o1);
    o2 = WMMA_BF16(pfrag, vb2, o2);
    o3 = WMMA_BF16(pfrag, vb3, o3);
    __syncthreads();
  }

  // ---- normalize and store bf16 attention output ----
  const size_t orow0 = ((size_t)(b * kN) + q0 + 8 * kh) * kDim + h * kHd + mr;
#pragma unroll
  for (int r = 0; r < 8; ++r) {
    const float inv = 1.0f / rsum[r];
    const size_t p = orow0 + (size_t)r * kDim;
    O[p]      = (bf16)(o0[r] * inv);
    O[p + 16] = (bf16)(o1[r] * inv);
    O[p + 32] = (bf16)(o2[r] * inv);
    O[p + 48] = (bf16)(o3[r] * inv);
  }
}

// ---------------- host side ----------------
extern "C" void kernel_launch(void* const* d_in, const int* in_sizes, int n_in,
                              void* d_out, int out_size, void* d_ws, size_t ws_size,
                              hipStream_t stream) {
  (void)in_sizes; (void)n_in; (void)out_size; (void)ws_size;
  const float* x   = (const float*)d_in[0];
  const float* Wq  = (const float*)d_in[1];
  const float* Wkv = (const float*)d_in[2];
  const float* Wp  = (const float*)d_in[3];
  float* out = (float*)d_out;

  char* p = (char*)d_ws;
  auto take = [&](size_t bytes) -> void* {
    void* r = p;
    p += (bytes + 255) & ~(size_t)255;
    return r;
  };
  bf16* x_bf   = (bf16*)take((size_t)kM * kDim * 2);
  bf16* wq_bf  = (bf16*)take((size_t)kDim * kDim * 2);
  bf16* wkv_bf = (bf16*)take((size_t)kKvDim * kDim * 2);
  bf16* wp_bf  = (bf16*)take((size_t)kDim * kDim * 2);
  bf16* q_bf   = (bf16*)take((size_t)kM * kDim * 2);
  bf16* kv_bf  = (bf16*)take((size_t)kM * kKvDim * 2);
  bf16* ao_bf  = (bf16*)take((size_t)kM * kDim * 2);

  auto cast = [&](const float* src, bf16* dst, size_t n) {
    int n4 = (int)(n / 4);
    cast_f32_to_bf16<<<(n4 + 255) / 256, 256, 0, stream>>>(src, dst, n4);
  };
  cast(x,   x_bf,   (size_t)kM * kDim);
  cast(Wq,  wq_bf,  (size_t)kDim * kDim);
  cast(Wkv, wkv_bf, (size_t)kKvDim * kDim);
  cast(Wp,  wp_bf,  (size_t)kDim * kDim);

  gemm_xwt<false><<<dim3(kDim / 64,   kM / 128), 256, 0, stream>>>(
      x_bf, wq_bf, nullptr, q_bf, kM, kDim, kDim);
  gemm_xwt<false><<<dim3(kKvDim / 64, kM / 128), 256, 0, stream>>>(
      x_bf, wkv_bf, nullptr, kv_bf, kM, kKvDim, kDim);

  attn_fwd<<<dim3(kN / 64, kHeads, kB), 128, 0, stream>>>(q_bf, kv_bf, ao_bf);

  gemm_xwt<true><<<dim3(kDim / 64, kM / 128), 256, 0, stream>>>(
      ao_bf, wp_bf, out, nullptr, kM, kDim, kDim);
}
